// SpatialTransformer_57664230916576
// MI455X (gfx1250) — matrix operand order repaired and checked by reading, not compiled, not run on hardware
//
#include <hip/hip_runtime.h>
#include <hip/hip_bf16.h>

typedef __attribute__((ext_vector_type(16))) __bf16 v16bf;
typedef __attribute__((ext_vector_type(8)))  __bf16 bf16x8;
typedef __attribute__((ext_vector_type(8)))  float  v8f;

constexpr int B_  = 8;
constexpr int N_  = 2048;
constexpr int ZD  = 512;
constexpr int SD  = 64;
constexpr int MLP = 2048;
constexpr int DEPTH = 4;
constexpr int T_  = B_ * N_;   // 16384 tokens

// ---------------------------------------------------------------------------
// Feature probes (device pass only; host pass sees the plain-copy fallback)
// ---------------------------------------------------------------------------
#if defined(__has_builtin)
#if __has_builtin(__builtin_amdgcn_tensor_load_to_lds)
#define USE_TDM 1
#else
#define USE_TDM 0
#endif
#if __has_builtin(__builtin_amdgcn_global_load_async_to_lds_b128)
#define USE_ASYNC 1
#else
#define USE_ASYNC 0
#endif
#else
#define USE_TDM 0
#define USE_ASYNC 0
#endif

#if USE_TDM
typedef __attribute__((ext_vector_type(4))) unsigned int u32x4;
typedef __attribute__((ext_vector_type(8))) int i32x8;
typedef __attribute__((ext_vector_type(4))) int i32x4;

__device__ inline unsigned int lds_off(const void* p) {
  // generic -> LDS(AS3) addrspacecast, then take the 32-bit LDS byte offset
  return (unsigned int)(unsigned long long)
      (__attribute__((address_space(3))) const void*)p;
}

// 2-D tile DMA: LDS[ldsoff..] <= tile1 rows x tile0 bf16 elements,
// source rows strided by `stride` elements, row length `rowlen` (OOB bound).
__device__ inline void tdm_load_2d(unsigned int ldsoff, const void* g,
                                   unsigned int rowlen, unsigned int tile0,
                                   unsigned int tile1,
                                   unsigned long long stride) {
  unsigned long long ga = (unsigned long long)g;
  u32x4 g0;
  g0[0] = 1u;                                            // count=1, user mode
  g0[1] = ldsoff;                                        // lds_addr
  g0[2] = (unsigned int)ga;                              // global_addr[31:0]
  g0[3] = (unsigned int)((ga >> 32) & 0x1FFFFFFull) | (2u << 30); // +type=2
  i32x8 g1;
  g1[0] = (int)(1u << 16);                               // data_size = 2 bytes
  g1[1] = (int)((rowlen & 0xffffu) << 16);               // tensor_dim0[15:0]
  g1[2] = (int)((rowlen >> 16) | ((tile1 & 0xffffu) << 16)); // dim0 hi, dim1 lo
  g1[3] = (int)((tile1 >> 16) | ((tile0 & 0xffffu) << 16));  // dim1 hi, tile0
  g1[4] = (int)(tile1 & 0xffffu);                        // tile_dim1 (tile2=0)
  g1[5] = (int)(unsigned int)(stride & 0xffffffffull);   // dim0_stride lo
  g1[6] = (int)(unsigned int)((stride >> 32) & 0xffffull);
  g1[7] = 0;
  i32x4 z4 = {0, 0, 0, 0};
  i32x8 z8 = {0, 0, 0, 0, 0, 0, 0, 0};
  // 6-arg form (clang-23 / therock HIP headers)
  __builtin_amdgcn_tensor_load_to_lds(g0, g1, z4, z4, z8, 0);
}
#endif

// 16-byte global -> LDS copy (async if available, else through VGPRs)
__device__ inline void copy16(__bf16* l, const __bf16* g) {
#if USE_ASYNC
  typedef int v4i_gcc __attribute__((__vector_size__(16)));
  __builtin_amdgcn_global_load_async_to_lds_b128(
      (__attribute__((address_space(1))) v4i_gcc*)g,
      (__attribute__((address_space(3))) v4i_gcc*)l, 0, 0);
#else
  *(bf16x8*)l = *(const bf16x8*)g;
#endif
}

// ---------------------------------------------------------------------------
// WMMA helpers (CDNA5 v_wmma_f32_16x16x32_bf16, wave32)
// ---------------------------------------------------------------------------
__device__ inline v8f wmma_bf16(v16bf a, v16bf b, v8f c) {
  return __builtin_amdgcn_wmma_f32_16x16x32_bf16(false, a, false, b, (short)0, c,
                                                 false, false);
}

// A fragment, 16x32 bf16, row-major K-contiguous (row stride ld elements).
__device__ inline v16bf load_frag_a(const __bf16* src, int ld) {
  const int l  = threadIdx.x & 31;
  const int m  = l & 15;
  const int hi = l >> 4;
  const __bf16* p = src + (size_t)m * ld + hi * 8;
  bf16x8 x0 = *(const bf16x8*)(p);
  bf16x8 x1 = *(const bf16x8*)(p + 16);
  v16bf r;
#pragma unroll
  for (int i = 0; i < 8; ++i) { r[i] = x0[i]; r[i + 8] = x1[i]; }
  return r;
}

// B fragment, 32x16 bf16, from K-contiguous [n][k] storage (row stride ld).
__device__ inline v16bf load_frag_b(const __bf16* src, int ld) {
  const int l  = threadIdx.x & 31;
  const int n  = l & 15;
  const int hi = l >> 4;
  const __bf16* p = src + (size_t)n * ld + hi * 8;
  bf16x8 x0 = *(const bf16x8*)(p);
  bf16x8 x1 = *(const bf16x8*)(p + 16);
  v16bf r;
#pragma unroll
  for (int i = 0; i < 8; ++i) { r[i] = x0[i]; r[i + 8] = x1[i]; }
  return r;
}

// ---------------------------------------------------------------------------
// Utility kernels
// ---------------------------------------------------------------------------
__global__ void cast_bf16_kernel(const float* __restrict__ x,
                                 __bf16* __restrict__ y, size_t n) {
  size_t i = (size_t)blockIdx.x * blockDim.x + threadIdx.x;
  size_t stride = (size_t)gridDim.x * blockDim.x;
  for (; i < n; i += stride) y[i] = (__bf16)x[i];
}

// f32 [R][C] -> bf16 [C][R] (stacked matrices on grid.z), 32x32 LDS tiles.
__global__ __launch_bounds__(256)
void transpose_cast_kernel(const float* __restrict__ x, __bf16* __restrict__ y,
                           int R, int C) {
  __shared__ float tile[32][33];
  const float* xm = x + (size_t)blockIdx.z * R * C;
  __bf16* ym = y + (size_t)blockIdx.z * R * C;
  const int c0 = blockIdx.x * 32, r0 = blockIdx.y * 32;
  const int tx = threadIdx.x & 31, ty = threadIdx.x >> 5;  // 32 x 8
#pragma unroll
  for (int i = 0; i < 32; i += 8)
    tile[ty + i][tx] = xm[(size_t)(r0 + ty + i) * C + c0 + tx];
  __syncthreads();
#pragma unroll
  for (int i = 0; i < 32; i += 8)
    ym[(size_t)(c0 + ty + i) * R + r0 + tx] = (__bf16)tile[tx][ty + i];
}

__global__ void sumsq_kernel(const float* __restrict__ s,
                             float* __restrict__ sq) {
  int t = blockIdx.x * blockDim.x + threadIdx.x;
  if (t >= T_) return;
  const float* p = s + (size_t)t * SD;
  float a = 0.f;
#pragma unroll
  for (int i = 0; i < SD; ++i) a += p[i] * p[i];
  sq[t] = a;
}

// LayerNorm over ZD=512 (one wave per token), fused cast to bf16.
__global__ __launch_bounds__(256)
void ln_cast_kernel(const float* __restrict__ x, const float* __restrict__ g,
                    const float* __restrict__ b, __bf16* __restrict__ y) {
  const int wave = threadIdx.x >> 5, lane = threadIdx.x & 31;
  const size_t t = (size_t)blockIdx.x * 8 + wave;
  const float* xr = x + t * ZD;
  float v[16];
  float s = 0.f;
#pragma unroll
  for (int i = 0; i < 16; ++i) { v[i] = xr[lane + i * 32]; s += v[i]; }
#pragma unroll
  for (int o = 16; o > 0; o >>= 1) s += __shfl_xor(s, o, 32);
  const float mean = s * (1.f / ZD);
  float var = 0.f;
#pragma unroll
  for (int i = 0; i < 16; ++i) { float d = v[i] - mean; var += d * d; }
#pragma unroll
  for (int o = 16; o > 0; o >>= 1) var += __shfl_xor(var, o, 32);
  const float inv = rsqrtf(var * (1.f / ZD) + 1e-5f);
#pragma unroll
  for (int i = 0; i < 16; ++i) {
    const int c = lane + i * 32;
    y[t * ZD + c] = (__bf16)((v[i] - mean) * inv * g[c] + b[c]);
  }
}

// ---------------------------------------------------------------------------
// Spatial attention: z_out = softmax(-max(d2,0)) @ z
// Block = 16 query rows of one batch. zt is V pre-transposed: [b][d][n] bf16.
// ---------------------------------------------------------------------------
__global__ __launch_bounds__(256)
void attn_kernel(const __bf16* __restrict__ eb, const float* __restrict__ sq,
                 const __bf16* __restrict__ zt, float* __restrict__ out) {
  __shared__ __align__(16) __bf16 Sbuf[16][N_];      // 64 KB: scores then P
  __shared__ __align__(16) __bf16 VT[2][ZD][32];     // 64 KB: V k-tiles [d][j]
  __shared__ float red[16][16];
  __shared__ float rowstat[16];
  __shared__ float s_sqi[16];

  const int tid = threadIdx.x, wave = tid >> 5, lane = tid & 31;
  const int b = blockIdx.y;
  const int i0 = blockIdx.x * 16;
  const int tbase = b * N_;
  const __bf16* eB = eb + (size_t)tbase * SD;
  const __bf16* ztB = zt + (size_t)b * ZD * N_;

  if (tid < 16) s_sqi[tid] = sq[tbase + i0 + tid];
  __syncthreads();

  v16bf qa0 = load_frag_a(eB + (size_t)i0 * SD + 0, SD);
  v16bf qa1 = load_frag_a(eB + (size_t)i0 * SD + 32, SD);

  const int nl = lane & 15, hib = lane >> 4;
  const v8f zerov = {0.f, 0.f, 0.f, 0.f, 0.f, 0.f, 0.f, 0.f};

  // ---- Phase 1: S[i,j] = min(2 e_i.e_j - |e_i|^2 - |e_j|^2, 0)
  for (int jt = wave * 256; jt < wave * 256 + 256; jt += 16) {
    v8f acc = zerov;
    acc = wmma_bf16(qa0, load_frag_b(eB + (size_t)jt * SD + 0, SD), acc);
    acc = wmma_bf16(qa1, load_frag_b(eB + (size_t)jt * SD + 32, SD), acc);
    const float sqj = sq[tbase + jt + nl];
#pragma unroll
    for (int r = 0; r < 8; ++r) {
      const int m = r + 8 * hib;
      Sbuf[m][jt + nl] = (__bf16)fminf(2.f * acc[r] - s_sqi[m] - sqj, 0.f);
    }
  }
  __syncthreads();

  // ---- Phase 2: row softmax (P written back into Sbuf as bf16)
  {
    const int row = tid >> 4, col = tid & 15;
    float mx = -3.0e38f;
    for (int j = col; j < N_; j += 16) mx = fmaxf(mx, (float)Sbuf[row][j]);
    red[row][col] = mx;
    __syncthreads();
    if (col == 0) {
      float m = red[row][0];
      for (int c = 1; c < 16; ++c) m = fmaxf(m, red[row][c]);
      rowstat[row] = m;
    }
    __syncthreads();
    const float rm = rowstat[row];
    float sm = 0.f;
    for (int j = col; j < N_; j += 16) {
      const float p = __expf((float)Sbuf[row][j] - rm);
      Sbuf[row][j] = (__bf16)p;
      sm += p;
    }
    __syncthreads();
    red[row][col] = sm;
    __syncthreads();
    if (col == 0) {
      float s2 = 0.f;
      for (int c = 0; c < 16; ++c) s2 += red[row][c];
      rowstat[row] = s2;
    }
    __syncthreads();
  }

  // ---- Phase 3: O = P @ V, V k-tile = 512 rows x 32 cols of zt (row-major)
  auto stageV = [&](int buf, int k0) {
#if USE_TDM
    if (wave == 0)
      tdm_load_2d(lds_off(&VT[buf][0][0]), ztB + k0, N_, 32, ZD, N_);
#else
    for (int g = tid; g < ZD * 4; g += 256) {
      const int d = g >> 2, jj = (g & 3) * 8;
      copy16(&VT[buf][d][jj], ztB + (size_t)d * N_ + k0 + jj);
    }
#endif
  };
  auto stage_sync = [&]() {
#if USE_TDM
    if (wave == 0) __builtin_amdgcn_s_wait_tensorcnt(0);
#else
#if USE_ASYNC
    asm volatile("s_wait_asynccnt 0x0" ::: "memory");
#endif
#endif
    __syncthreads();
  };

  v8f oacc[4] = {zerov, zerov, zerov, zerov};
  const int d0 = wave * 64;
  constexpr int VSTEPS = N_ / 32;
  stageV(0, 0);
  stage_sync();
  for (int kt = 0; kt < VSTEPS; ++kt) {
    const int buf = kt & 1;
    if (kt + 1 < VSTEPS) stageV(buf ^ 1, (kt + 1) * 32);
    v16bf pa = load_frag_a(&Sbuf[0][kt * 32], N_);
#pragma unroll
    for (int nb = 0; nb < 4; ++nb) {
      v16bf bb = load_frag_b(&VT[buf][d0 + nb * 16][0], 32);
      oacc[nb] = wmma_bf16(pa, bb, oacc[nb]);
    }
    stage_sync();
  }
#pragma unroll
  for (int nb = 0; nb < 4; ++nb) {
#pragma unroll
    for (int r = 0; r < 8; ++r) {
      const int m = r + 8 * hib;
      out[(size_t)(tbase + i0 + m) * ZD + d0 + nb * 16 + nl] =
          oacc[nb][r] / rowstat[m];
    }
  }
}

// ---------------------------------------------------------------------------
// Tiled bf16 GEMM, TDM-staged, double buffered, k-step 64.
//   A:  [Mtot][KD] bf16 row-major.   Bt: [Nall][KD] bf16 (pre-transposed W).
//   out = op(A @ Bt^T + bias [+ resid]); 8 waves, each 32x64 per block tile.
// ---------------------------------------------------------------------------
template <int MT, int NT, int KD, bool GELU, bool RES, bool OBF>
__global__ __launch_bounds__(256)
void gemm_bf16_kernel(const __bf16* __restrict__ A,
                      const __bf16* __restrict__ Bt,
                      const float* __restrict__ bias,
                      const float* __restrict__ resid,
                      float* __restrict__ outf, __bf16* __restrict__ outb,
                      int Nall) {
  constexpr int KS = 64;                     // staged k-tile width
  __shared__ __align__(16) __bf16 As[2][MT][KS];
  __shared__ __align__(16) __bf16 Bs[2][NT][KS];

  constexpr int WN = 64;
  constexpr int WAVES_N = NT / WN;
  constexpr int WAVES_M = 8 / WAVES_N;
  static_assert(MT / WAVES_M == 32, "wave tile must be 32 rows");

  const int tid = threadIdx.x, wave = tid >> 5, lane = tid & 31;
  const int wm = wave % WAVES_M, wn = wave / WAVES_M;
  const size_t m0 = (size_t)blockIdx.x * MT;
  const size_t n0 = (size_t)blockIdx.y * NT;

  auto stage = [&](int buf, int k0) {
#if USE_TDM
    if (wave == 0) {
      tdm_load_2d(lds_off(&As[buf][0][0]), A + m0 * KD + k0, KD, KS, MT, KD);
      tdm_load_2d(lds_off(&Bs[buf][0][0]), Bt + n0 * KD + k0, KD, KS, NT, KD);
    }
#else
    for (int g = tid; g < MT * (KS / 8); g += 256) {
      const int m = g / (KS / 8), kk = (g % (KS / 8)) * 8;
      copy16(&As[buf][m][kk], A + (m0 + m) * KD + k0 + kk);
    }
    for (int g = tid; g < NT * (KS / 8); g += 256) {
      const int n = g / (KS / 8), kk = (g % (KS / 8)) * 8;
      copy16(&Bs[buf][n][kk], Bt + (n0 + n) * KD + k0 + kk);
    }
#endif
  };
  auto stage_sync = [&]() {
#if USE_TDM
    if (wave == 0) __builtin_amdgcn_s_wait_tensorcnt(0);
#else
#if USE_ASYNC
    asm volatile("s_wait_asynccnt 0x0" ::: "memory");
#endif
#endif
    __syncthreads();
  };

  const v8f zerov = {0.f, 0.f, 0.f, 0.f, 0.f, 0.f, 0.f, 0.f};
  v8f acc[2][4];
#pragma unroll
  for (int mb = 0; mb < 2; ++mb)
#pragma unroll
    for (int nb = 0; nb < 4; ++nb) acc[mb][nb] = zerov;

  constexpr int STEPS = KD / KS;
  stage(0, 0);
  stage_sync();
  for (int kt = 0; kt < STEPS; ++kt) {
    const int buf = kt & 1;
    if (kt + 1 < STEPS) stage(buf ^ 1, (kt + 1) * KS);
#pragma unroll
    for (int kk = 0; kk < KS; kk += 32) {
      v16bf af0 = load_frag_a(&As[buf][wm * 32][kk], KS);
      v16bf af1 = load_frag_a(&As[buf][wm * 32 + 16][kk], KS);
#pragma unroll
      for (int nb = 0; nb < 4; ++nb) {
        v16bf bb = load_frag_b(&Bs[buf][wn * 64 + nb * 16][kk], KS);
        acc[0][nb] = wmma_bf16(af0, bb, acc[0][nb]);
        acc[1][nb] = wmma_bf16(af1, bb, acc[1][nb]);
      }
    }
    stage_sync();
  }

  const int nl = lane & 15, hib = lane >> 4;
#pragma unroll
  for (int mb = 0; mb < 2; ++mb)
#pragma unroll
    for (int nb = 0; nb < 4; ++nb)
#pragma unroll
      for (int r = 0; r < 8; ++r) {
        const size_t m = m0 + wm * 32 + mb * 16 + r + 8 * hib;
        const int n = (int)n0 + wn * 64 + nb * 16 + nl;
        float v = acc[mb][nb][r] + bias[n];
        if (GELU) v = 0.5f * v * (1.f + erff(v * 0.70710678f));
        if (RES) v += resid[m * Nall + n];
        if (OBF) outb[m * Nall + n] = (__bf16)v;
        else     outf[m * Nall + n] = v;
      }
}

// ---------------------------------------------------------------------------
// Host orchestration
// ---------------------------------------------------------------------------
extern "C" void kernel_launch(void* const* d_in, const int* in_sizes, int n_in,
                              void* d_out, int out_size, void* d_ws,
                              size_t ws_size, hipStream_t stream) {
  const float* z_in  = (const float*)d_in[0];
  const float* s_in0 = (const float*)d_in[1];
  const float* lnzs  = (const float*)d_in[2];
  const float* lnzb  = (const float*)d_in[3];
  const float* lnss  = (const float*)d_in[4];
  const float* lnsb  = (const float*)d_in[5];
  const float* Wz1   = (const float*)d_in[6];
  const float* bz1   = (const float*)d_in[7];
  const float* Wz2   = (const float*)d_in[8];
  const float* bz2   = (const float*)d_in[9];
  const float* Ws1   = (const float*)d_in[10];
  const float* bs1   = (const float*)d_in[11];
  const float* Ws2   = (const float*)d_in[12];
  const float* bs2   = (const float*)d_in[13];

  char* ws = (char*)d_ws;
  size_t off = 0;
  auto alloc = [&](size_t bytes) -> char* {
    char* p = ws + off;
    off += (bytes + 255) & ~(size_t)255;
    return p;
  };
  // weights pre-transposed to [N][K] bf16
  __bf16* wz1t = (__bf16*)alloc((size_t)DEPTH * ZD * MLP * 2);
  __bf16* wz2t = (__bf16*)alloc((size_t)DEPTH * MLP * ZD * 2);
  __bf16* ws1t = (__bf16*)alloc((size_t)DEPTH * ZD * MLP * 2);
  __bf16* ws2t = (__bf16*)alloc((size_t)DEPTH * MLP * SD * 2);
  float* zcur  = (float*)alloc((size_t)T_ * ZD * 4);
  float* scur  = (float*)alloc((size_t)T_ * SD * 4);
  float* zatt  = (float*)alloc((size_t)T_ * ZD * 4);
  __bf16* zt   = (__bf16*)alloc((size_t)T_ * ZD * 2);  // per-batch [d][n]
  __bf16* ebuf = (__bf16*)alloc((size_t)T_ * SD * 2);
  float* sqv   = (float*)alloc((size_t)T_ * 4);
  __bf16* lnb  = (__bf16*)alloc((size_t)T_ * ZD * 2);
  __bf16* Hb   = (__bf16*)alloc((size_t)T_ * MLP * 2);

  // weights -> transposed bf16 (every call; deterministic)
  transpose_cast_kernel<<<dim3(MLP/32, ZD/32, DEPTH), 256, 0, stream>>>(Wz1, wz1t, ZD, MLP);
  transpose_cast_kernel<<<dim3(ZD/32, MLP/32, DEPTH), 256, 0, stream>>>(Wz2, wz2t, MLP, ZD);
  transpose_cast_kernel<<<dim3(MLP/32, ZD/32, DEPTH), 256, 0, stream>>>(Ws1, ws1t, ZD, MLP);
  transpose_cast_kernel<<<dim3(SD/32, MLP/32, DEPTH), 256, 0, stream>>>(Ws2, ws2t, MLP, SD);

  float* outz = (float*)d_out;
  float* outs = outz + (size_t)T_ * ZD;

  for (int i = 0; i < DEPTH; ++i) {
    const float* zin = (i == 0) ? z_in : zcur;
    const float* sin = (i == 0) ? s_in0 : scur;
    float* zout = (i == DEPTH - 1) ? outz : zcur;
    float* sout = (i == DEPTH - 1) ? outs : scur;

    cast_bf16_kernel<<<256, 256, 0, stream>>>(sin, ebuf, (size_t)T_ * SD);
    sumsq_kernel<<<T_ / 256, 256, 0, stream>>>(sin, sqv);
    // z -> per-batch transposed bf16 (V for attention)
    transpose_cast_kernel<<<dim3(ZD/32, N_/32, B_), 256, 0, stream>>>(zin, zt, N_, ZD);

    attn_kernel<<<dim3(N_ / 16, B_), 256, 0, stream>>>(ebuf, sqv, zt, zatt);

    ln_cast_kernel<<<T_ / 8, 256, 0, stream>>>(zatt, lnzs + i * ZD,
                                               lnzb + i * ZD, lnb);
    gemm_bf16_kernel<128, 128, ZD, true, false, true>
        <<<dim3(T_ / 128, MLP / 128), 256, 0, stream>>>(
            lnb, wz1t + (size_t)i * ZD * MLP, bz1 + i * MLP, nullptr, nullptr,
            Hb, MLP);
    gemm_bf16_kernel<128, 128, MLP, false, true, false>
        <<<dim3(T_ / 128, ZD / 128), 256, 0, stream>>>(
            Hb, wz2t + (size_t)i * MLP * ZD, bz2 + i * ZD, zatt, zout, nullptr,
            ZD);

    ln_cast_kernel<<<T_ / 8, 256, 0, stream>>>(zout, lnss + i * ZD,
                                               lnsb + i * ZD, lnb);
    gemm_bf16_kernel<128, 128, ZD, true, false, true>
        <<<dim3(T_ / 128, MLP / 128), 256, 0, stream>>>(
            lnb, ws1t + (size_t)i * ZD * MLP, bs1 + i * MLP, nullptr, nullptr,
            Hb, MLP);
    gemm_bf16_kernel<256, 64, MLP, false, true, false>
        <<<dim3(T_ / 256, SD / 64), 256, 0, stream>>>(
            Hb, ws2t + (size_t)i * MLP * SD, bs2 + i * SD, sin, sout, nullptr,
            SD);
  }

  (void)in_sizes; (void)n_in; (void)out_size; (void)ws_size;
}
